// PICNNEntropicOTQuantileRegression_59184649339523
// MI455X (gfx1250) — compile-verified, matrix operand order checked by reading, not run on hardware
//
#include <hip/hip_runtime.h>
#include <hip/hip_bf16.h>
#include <math.h>

typedef __bf16 bf16_t;
typedef __attribute__((ext_vector_type(16))) __bf16 v16bf;
typedef __attribute__((ext_vector_type(8)))  float  v8f;
typedef __attribute__((ext_vector_type(4)))  unsigned int u32x4;
typedef __attribute__((ext_vector_type(4)))  int i32x4;
typedef __attribute__((ext_vector_type(8)))  int i32x8;

#define NN 128
#define MM 2048
#define DX 64
#define DY 8
#define DU 256
#define DZ 256
#define EPSV 0.01f

// ---- workspace layout (bytes) ----
// [0, 147456)          : B_main bf16, fragment-ordered  (9 zsteps x 16 ktiles x 32 lanes x 16)
// [147456, ...)        : per-n records, 1056 floats each:
//    [0..7] gy0  [8..15] gy1  [16..23] yv  [24] c2  [25..31] pad
//    [32..287] gz1  [288..543] Xu0  [544..799] uW1  [800..1055] g2
#define REC_STRIDE 1056
#define BMAIN_BYTES 147456

__device__ __forceinline__ float softplusf(float x){
  return x > 20.f ? x : log1pf(__expf(x));
}
// A/B 16-bit fragment element pattern: element j of the lane's 16 values maps to
// K = ((j>>1)&3)*2 + (j&1) + (j>>3)*16 + ((lane>>4)&1)*8   (per ISA 16-bit A 16x32 table)
__device__ __forceinline__ int kk_pat(int j, int lane){
  return (((j>>1)&3)<<1) + (j&1) + ((j>>3)<<4) + (((lane>>4)&1)<<3);
}

// ---------------- prep 1: global B matrix for the big GEMM (n-independent) ----------------
// B[zK, k]: zK<256 -> softplus(Wz1[k][zK]); 256<=zK<264 -> Wy1[k][zK-256]; else 0
__global__ void __launch_bounds__(256) prep_bmain(const float* __restrict__ Wz1,
                                                  const float* __restrict__ Wy1,
                                                  bf16_t* __restrict__ bmain){
  int flat = blockIdx.x * 256 + threadIdx.x;        // 288 blocks -> 73728 elements
  int j    = flat & 15;
  int ln   = (flat >> 4) & 31;
  int t    = (flat >> 9) & 15;
  int s    = flat >> 13;                            // 0..8
  int k    = t * 16 + (ln & 15);
  int zK   = s * 32 + kk_pat(j, ln);
  float v = 0.f;
  if (zK < 256)       v = softplusf(Wz1[k * DZ + zK]);
  else if (zK < 264)  v = Wy1[k * DY + (zK - 256)];
  bmain[flat] = (bf16_t)v;
}

// ---------------- prep 2: per-n context vectors ----------------
__global__ void __launch_bounds__(256) prep_ctx(
    const float* __restrict__ X,   const float* __restrict__ Y,
    const float* __restrict__ Wt0, const float* __restrict__ bt0,
    const float* __restrict__ Wt1, const float* __restrict__ bt1,
    const float* __restrict__ Wyu0,const float* __restrict__ byu0,
    const float* __restrict__ Wu0, const float* __restrict__ b0,
    const float* __restrict__ Wzu1,const float* __restrict__ bzu1,
    const float* __restrict__ Wyu1,const float* __restrict__ byu1,
    const float* __restrict__ Wu1, const float* __restrict__ b1,
    const float* __restrict__ Wzu2,const float* __restrict__ bzu2,
    const float* __restrict__ Wz2,
    const float* __restrict__ Wyu2,const float* __restrict__ byu2,
    const float* __restrict__ Wy2, const float* __restrict__ Wu2,
    const float* __restrict__ b2,
    float* __restrict__ rec){
  __shared__ float Xs[DX], u1s[DU], u2s[DU], red[256];
  const int n = blockIdx.x, j = threadIdx.x;
  float* R = rec + n * REC_STRIDE;
  if (j < DX) Xs[j] = X[n * DX + j];
  __syncthreads();
  float a = bt0[j];
  for (int x = 0; x < DX; x++) a += Xs[x] * Wt0[j * DX + x];
  u1s[j] = fmaxf(a, 0.f);
  __syncthreads();
  float bb = bt1[j];
  for (int x = 0; x < DU; x++) bb += u1s[x] * Wt1[j * DU + x];
  u2s[j] = fmaxf(bb, 0.f);
  __syncthreads();
  float gz1 = bzu1[j], uw = b1[j], gz2 = bzu2[j];
  for (int x = 0; x < DU; x++){
    float u1x = u1s[x], u2x = u2s[x];
    gz1 += u1x * Wzu1[j * DU + x];
    uw  += u1x * Wu1 [j * DU + x];
    gz2 += u2x * Wzu2[j * DU + x];
  }
  R[ 32 + j] = fmaxf(gz1, 0.f);
  R[544 + j] = uw;
  R[800 + j] = fmaxf(gz2, 0.f) * softplusf(Wz2[j]);
  float xu = b0[j];
  for (int x = 0; x < DX; x++) xu += Xs[x] * Wu0[j * DX + x];
  R[288 + j] = xu;
  if (j < DY){
    float g0 = byu0[j];
    for (int x = 0; x < DX; x++) g0 += Xs[x] * Wyu0[j * DX + x];
    float g1 = byu1[j], g2y = byu2[j];
    for (int x = 0; x < DU; x++){
      g1  += u1s[x] * Wyu1[j * DU + x];
      g2y += u2s[x] * Wyu2[j * DU + x];
    }
    R[j]      = g0;                            // gy0
    R[8 + j]  = g1;                            // gy1
    R[16 + j] = Y[n * DY + j] - g2y * Wy2[j];  // yv = Y - gy2*Wy2
  }
  red[j] = u2s[j] * Wu2[j];
  __syncthreads();
  for (int off = 128; off > 0; off >>= 1){
    if (j < off) red[j] += red[j + off];
    __syncthreads();
  }
  if (j == 0) R[24] = red[0] + b2[0];          // c2
}

// ---------------- main fused kernel: one WG per n ----------------
// dynamic LDS carve (bytes):
#define L_BM   0         // 147456  B_main bf16
#define L_B0   147456    // 16384   z1-GEMM B (per-n) bf16
#define L_AW   163840    // 73728   A fragments: 8 waves x 9 zsteps x 32 lanes x 16 bf16
#define L_UL   237568    // 4096    U tile [128][8] f32
#define L_UY   241664    // 512     U . yv  [128] f32
#define L_UW   242176    // 1024    uW1 [256]
#define L_G2   243200    // 1024    g2  [256]
#define L_SV   244224    // 128     gy0/gy1/yv/c2
#define L_RD   244352    // 2048    lse reduce [256][2]
#define LDS_TOTAL 246400

__global__ void __launch_bounds__(256) picnn_main(
    const float*  __restrict__ Uin,
    const float*  __restrict__ Wy0,
    const bf16_t* __restrict__ bmain_g,
    const float*  __restrict__ rec,
    float*        __restrict__ out){
  extern __shared__ char smraw[];
  bf16_t* BM = (bf16_t*)(smraw + L_BM);
  bf16_t* B0 = (bf16_t*)(smraw + L_B0);
  bf16_t* AWall = (bf16_t*)(smraw + L_AW);
  float* UL = (float*)(smraw + L_UL);
  float* UY = (float*)(smraw + L_UY);
  float* UW = (float*)(smraw + L_UW);
  float* G2 = (float*)(smraw + L_G2);
  float* SV = (float*)(smraw + L_SV);
  float* RD = (float*)(smraw + L_RD);

  const int n = blockIdx.x, tid = threadIdx.x;
  const int wave = tid >> 5, lane = tid & 31;
  const float* R = rec + n * REC_STRIDE;
  bf16_t* AW = AWall + wave * (9 * 32 * 16);

  // ---- stage n-independent B_main into LDS via the Tensor Data Mover ----
#if __has_builtin(__builtin_amdgcn_tensor_load_to_lds)
  if (wave == 0){
    // D# group 0: count=1 | lds_addr | global_addr(57b) | type=2
    unsigned long long ga = (unsigned long long)(uintptr_t)bmain_g;
    unsigned int lds_off = (unsigned int)(uintptr_t)BM;   // low 32 bits of LDS aperture addr
    u32x4 g0;
    g0[0] = 1u;                                   // count=1, is_restore=0, gather=0
    g0[1] = lds_off;                              // lds_addr
    g0[2] = (unsigned int)(ga & 0xffffffffu);     // global_addr[31:0]
    g0[3] = (unsigned int)((ga >> 32) & 0x1ffffffu) | (2u << 30);  // addr[56:32] | type=2
    // D# group 1: 1-D tile of 18432 x 8B elements (= 147456 B)
    i32x8 g1;
    g1[0] = (3 << 16);            // workgroup_mask=0, data_size=3 (8B)
    g1[1] = (int)(18432u << 16);  // tensor_dim0[15:0] in [31:16]
    g1[2] = (1 << 16);            // tensor_dim0[31:16]=0, tensor_dim1=1
    g1[3] = (int)(18432u << 16);  // tile_dim0 = 18432
    g1[4] = 0;                    // tile_dim1=0 (unused), tile_dim2=0
    g1[5] = 18432;                // tensor_dim0_stride[31:0]
    g1[6] = 0;                    // stride hi, tensor_dim1_stride lo
    g1[7] = 0;
    i32x4 gz = {0, 0, 0, 0};
#if __clang_major__ >= 23
    i32x8 gz8 = {0, 0, 0, 0, 0, 0, 0, 0};
    __builtin_amdgcn_tensor_load_to_lds(g0, g1, gz, gz, gz8, 0);
#else
    __builtin_amdgcn_tensor_load_to_lds(g0, g1, gz, gz, 0);
#endif
    __builtin_amdgcn_s_wait_tensorcnt(0);
  }
#else
  {
    const uint4* src = (const uint4*)bmain_g;
    uint4* dst = (uint4*)BM;
    for (int i = tid; i < BMAIN_BYTES / 16; i += 256) dst[i] = src[i];
  }
#endif

  UW[tid] = R[544 + tid];
  G2[tid] = R[800 + tid];
  if (tid < 25) SV[tid] = R[tid];
  // build per-n B0 for the z1 GEMM, fragment-ordered: gz1 folded in (gz1>=0)
  for (int i = tid; i < 8192; i += 256){
    int j = i & 15, ln = (i >> 4) & 31, t = i >> 9;
    int z  = t * 16 + (ln & 15);
    int kk = kk_pat(j, ln);
    float v = 0.f;
    if (kk < 8)       v = Wy0[z * DY + kk] * R[32 + z];
    else if (kk == 8) v = R[288 + z]       * R[32 + z];
    B0[i] = (bf16_t)v;
  }
  __syncthreads();

  float gy0s[8], gy1s[8], yvs[8];
  #pragma unroll
  for (int d = 0; d < 8; d++){ gy0s[d] = SV[d]; gy1s[d] = SV[8 + d]; yvs[d] = SV[16 + d]; }
  const float c2s = SV[24];

  float mx = -INFINITY, smv = 0.f;      // online logsumexp state (per lane)

  for (int pass = 0; pass < 16; pass++){
    __syncthreads();
    const int m0 = pass * 128;
    if (tid < 128){
      const float* up = Uin + (size_t)(m0 + tid) * DY;
      float s = 0.f;
      #pragma unroll
      for (int d = 0; d < 8; d++){ float u = up[d]; UL[tid * 8 + d] = u; s += u * yvs[d]; }
      UY[tid] = s;
    }
    __syncthreads();

    const int lb = wave * 16;               // this wave's pass-local row base
    const int mrow = lb + (lane & 15);
    // A0 fragment (shared across all 16 z-tiles): U*gy0 | 1 | 0...
    v16bf a0;
    #pragma unroll
    for (int j = 0; j < 16; j++){
      int kk = kk_pat(j, lane);
      float v = (kk < 8) ? UL[mrow * 8 + kk] * gy0s[kk] : (kk == 8 ? 1.f : 0.f);
      a0[j] = (bf16_t)v;
    }
    // z1 GEMM: 16 WMMAs -> relu -> scatter into A-fragment layout in LDS
    #pragma unroll 2
    for (int t = 0; t < 16; t++){
      v16bf b = *(const v16bf*)(B0 + (t * 32 + lane) * 16);
      v8f c = {};
      c = __builtin_amdgcn_wmma_f32_16x16x32_bf16(false, a0, false, b, (short)0, c, false, false);
      const int mloc0 = (lane >> 4) * 8;
      const int zcol  = t * 16 + (lane & 15);
      const int s_ = zcol >> 5, kk = zcol & 31;
      const int half = (kk >> 3) & 1;
      const int bb = kk - (half << 3);
      const int low = bb & 15, hi = bb >> 4;
      const int jj = (((hi << 2) + (low >> 1)) << 1) + (low & 1);
      #pragma unroll
      for (int r = 0; r < 8; r++){
        float v = fmaxf(c[r], 0.f);
        int laneA = (half << 4) + mloc0 + r;
        AW[(s_ * 32 + laneA) * 16 + jj] = (bf16_t)v;
      }
    }
    // zstep 8: K=256..263 -> U*gy1, rest zero  (for kk<8: laneA=M, jj=kk)
    #pragma unroll
    for (int j = 0; j < 16; j++){
      float v = (lane < 16 && j < 8) ? UL[(lb + lane) * 8 + j] * gy1s[j] : 0.f;
      AW[(8 * 32 + lane) * 16 + j] = (bf16_t)v;
    }
    __syncthreads();

    // main GEMM: 9 zsteps x 16 ktiles = 144 WMMAs per wave (A frag pipelined)
    v8f acc[16];
    #pragma unroll
    for (int t = 0; t < 16; t++) acc[t] = (v8f){0,0,0,0,0,0,0,0};
    v16bf af = *(const v16bf*)(AW + (0 * 32 + lane) * 16);
    for (int s = 0; s < 9; s++){
      v16bf afn;
      if (s < 8) afn = *(const v16bf*)(AW + ((s + 1) * 32 + lane) * 16);
      #pragma unroll
      for (int t = 0; t < 16; t++){
        v16bf bfr = *(const v16bf*)(BM + ((s * 16 + t) * 32 + lane) * 16);
        acc[t] = __builtin_amdgcn_wmma_f32_16x16x32_bf16(false, af, false, bfr, (short)0, acc[t], false, false);
      }
      af = afn;
    }
    // epilogue: z2 = relu(acc + uW1[k]); phi-dot with g2
    float dotp[8];
    #pragma unroll
    for (int r = 0; r < 8; r++) dotp[r] = 0.f;
    #pragma unroll
    for (int t = 0; t < 16; t++){
      int kc = t * 16 + (lane & 15);
      float uw = UW[kc], gg = G2[kc];
      #pragma unroll
      for (int r = 0; r < 8; r++){
        float z2 = fmaxf(acc[t][r] + uw, 0.f);
        dotp[r] += z2 * gg;
      }
    }
    #pragma unroll
    for (int r = 0; r < 8; r++){
      float v = dotp[r];
      v += __shfl_xor(v, 1, 16);
      v += __shfl_xor(v, 2, 16);
      v += __shfl_xor(v, 4, 16);
      v += __shfl_xor(v, 8, 16);
      dotp[r] = v;
    }
    {
      const int rown = lane & 15;
      float slack = -INFINITY;
      if (rown < 8){
        int mloc = rown + ((lane >> 4) << 3);
        slack = UY[lb + mloc] - (dotp[rown] + c2s);
      }
      float nm = fmaxf(mx, slack);
      if (nm != -INFINITY){
        float e1 = __expf((mx - nm) * (1.f / EPSV));
        float e2 = (slack == -INFINITY) ? 0.f : __expf((slack - nm) * (1.f / EPSV));
        smv = smv * e1 + e2;
        mx = nm;
      }
    }
  }

  // block-wide online-LSE merge
  RD[tid * 2]     = mx;
  RD[tid * 2 + 1] = smv;
  __syncthreads();
  if (tid == 0){
    float MX = -INFINITY, SM = 0.f;
    for (int i = 0; i < 256; i++){
      float m_i = RD[i * 2], s_i = RD[i * 2 + 1];
      float nm = fmaxf(MX, m_i);
      if (nm == -INFINITY) continue;
      SM = SM * __expf((MX - nm) * (1.f / EPSV)) + s_i * __expf((m_i - nm) * (1.f / EPSV));
      MX = nm;
    }
    out[n] = EPSV * (logf(SM) - logf((float)MM)) + MX;
  }
}

extern "C" void kernel_launch(void* const* d_in, const int* in_sizes, int n_in,
                              void* d_out, int out_size, void* d_ws, size_t ws_size,
                              hipStream_t stream) {
  (void)in_sizes; (void)n_in; (void)out_size; (void)ws_size;
  const float* X    = (const float*)d_in[0];
  const float* U    = (const float*)d_in[1];
  const float* Y    = (const float*)d_in[2];
  const float* Wt0  = (const float*)d_in[3];
  const float* bt0  = (const float*)d_in[4];
  const float* Wt1  = (const float*)d_in[5];
  const float* bt1  = (const float*)d_in[6];
  const float* Wyu0 = (const float*)d_in[7];
  const float* byu0 = (const float*)d_in[8];
  const float* Wy0  = (const float*)d_in[9];
  const float* Wu0  = (const float*)d_in[10];
  const float* b0   = (const float*)d_in[11];
  const float* Wzu1 = (const float*)d_in[12];
  const float* bzu1 = (const float*)d_in[13];
  const float* Wz1  = (const float*)d_in[14];
  const float* Wyu1 = (const float*)d_in[15];
  const float* byu1 = (const float*)d_in[16];
  const float* Wy1  = (const float*)d_in[17];
  const float* Wu1  = (const float*)d_in[18];
  const float* b1   = (const float*)d_in[19];
  const float* Wzu2 = (const float*)d_in[20];
  const float* bzu2 = (const float*)d_in[21];
  const float* Wz2  = (const float*)d_in[22];
  const float* Wyu2 = (const float*)d_in[23];
  const float* byu2 = (const float*)d_in[24];
  const float* Wy2  = (const float*)d_in[25];
  const float* Wu2  = (const float*)d_in[26];
  const float* b2   = (const float*)d_in[27];

  bf16_t* bmain = (bf16_t*)d_ws;
  float*  rec   = (float*)((char*)d_ws + BMAIN_BYTES);

  // idempotent, called every launch (no static state)
  hipFuncSetAttribute((const void*)picnn_main,
                      hipFuncAttributeMaxDynamicSharedMemorySize, LDS_TOTAL);

  prep_bmain<<<288, 256, 0, stream>>>(Wz1, Wy1, bmain);
  prep_ctx<<<NN, 256, 0, stream>>>(X, Y, Wt0, bt0, Wt1, bt1,
                                   Wyu0, byu0, Wu0, b0,
                                   Wzu1, bzu1,
                                   Wyu1, byu1, Wu1, b1,
                                   Wzu2, bzu2, Wz2,
                                   Wyu2, byu2, Wy2, Wu2, b2, rec);
  picnn_main<<<NN, 256, LDS_TOTAL, stream>>>(U, Wy0, bmain, rec, (float*)d_out);
}
